// BatchRNN_55654186221574
// MI455X (gfx1250) — matrix operand order repaired and checked by reading, not compiled, not run on hardware
//
#include <hip/hip_runtime.h>
#include <cstdint>
#include <cstddef>

// ---------------------------------------------------------------------------
// Bidirectional masked LSTM for MI455X (gfx1250, wave32, WMMA + TDM).
//
// Recurrence: 1200 sequential steps of (32 x 1024 x 2048) GEMM per direction
// -> latency/L2 bound. Fused bf16 weights [W_ih|W_hh] (8 MB both dirs) stay
// resident in the 192 MB L2. GEMM flops via v_wmma_f32_16x16x32_bf16 (f32
// accumulate). Per step each block DMAs its 128 KB B working set into LDS
// with two tensor_load_to_lds ops (3D D#: x=512 elems, y=gate, z=k-tile),
// overlapping the second half's DMA with the first half's WMMAs; inner loop
// then reads B via short-latency ds_load_b128 instead of stalling on L2.
// ---------------------------------------------------------------------------

#define T_STEPS 1200
#define BATCH   32
#define IN      512
#define HID     512
#define G4H     2048      // 4*HID
#define KT      32        // number of K-tiles (of 32)
#define KTH     16        // K-tiles per half
#define NT      128       // number of N-tiles (of 16) per direction

typedef __attribute__((ext_vector_type(16))) __bf16 v16bf;
typedef __attribute__((ext_vector_type(8)))  __bf16 v8bf;
typedef __attribute__((ext_vector_type(8)))  float  v8f;
typedef __attribute__((ext_vector_type(4)))  uint32_t u32x4;
typedef __attribute__((ext_vector_type(8)))  int      i32x8;
typedef __attribute__((ext_vector_type(4)))  int      i32x4;

#if __has_builtin(__builtin_amdgcn_tensor_load_to_lds)
#define HAVE_TDM 1
#else
#define HAVE_TDM 0
#endif

static __device__ __forceinline__ float sigmoidf_(float x) {
  return 1.0f / (1.0f + __expf(-x));
}

static __device__ __forceinline__ v16bf cat16(v8bf lo, v8bf hi) {
  return __builtin_shufflevector(lo, hi, 0,1,2,3,4,5,6,7,8,9,10,11,12,13,14,15);
}

static __device__ __forceinline__ v8f wmma_bf16(v16bf a, v16bf b, v8f c) {
  // D = A(16x32 bf16) * B(32x16 bf16) + C(16x16 f32)
  return __builtin_amdgcn_wmma_f32_16x16x32_bf16(
      /*neg_a=*/false, a, /*neg_b=*/false, b,
      /*c_mod=*/(short)0, c, /*reuse_a=*/false, /*reuse_b=*/false);
}

#if HAVE_TDM
// One TDM 3D tile load: x = 512 bf16 (one fragment-packed 32x16 (k x n) tile),
// y = 4 gates (stride 32*512 elems), z = ktiles k-tiles (stride 128*512
// elems). Dest: contiguous LDS at lds_off. Fields per CDNA5 ISA section 8.
// This toolchain exposes the 6-arg builtin: (g0 u32x4, g1 i32x8, g2 i32x4,
// g3 i32x4, extra i32x8, cpol i32); the trailing i32x8 is zero-filled.
static __device__ __forceinline__ void tdm_load_B(const __bf16* gsrc,
                                                  uint32_t lds_off, int ktiles) {
  uint64_t ga = (uint64_t)(uintptr_t)gsrc;
  u32x4 g0;
  g0[0] = 1u;                                   // count=1, user mode
  g0[1] = lds_off;                              // lds_addr (bytes)
  g0[2] = (uint32_t)ga;                         // global_addr[31:0]
  g0[3] = (uint32_t)(ga >> 32) | (2u << 30);    // global_addr[56:32] | type=2
  i32x8 g1;
  g1[0] = 0x00010000;                           // data_size=1 (2 bytes)
  g1[1] = (int)(512u << 16);                    // tensor_dim0 = 512
  g1[2] = (int)(4u << 16);                      // tensor_dim1 = 4
  g1[3] = (int)(512u << 16);                    // tile_dim0 = 512
  g1[4] = (int)(((uint32_t)ktiles << 16) | 4u); // tile_dim1 = 4, tile_dim2
  g1[5] = 32 * 512;                             // tensor_dim0_stride = 16384
  g1[6] = 0;                                    // dim0_stride hi / dim1_stride lo
  g1[7] = 1;                                    // tensor_dim1_stride = 65536
  i32x4 g2; g2[0] = ktiles; g2[1] = 0; g2[2] = 0; g2[3] = 0;   // tensor_dim2
  i32x4 g3; g3[0] = 0; g3[1] = 0; g3[2] = 0; g3[3] = 0;
  i32x8 gz; gz[0] = 0; gz[1] = 0; gz[2] = 0; gz[3] = 0;
            gz[4] = 0; gz[5] = 0; gz[6] = 0; gz[7] = 0;
  __builtin_amdgcn_tensor_load_to_lds(g0, g1, g2, g3, gz, 0);
}
#endif

// ---------------------------------------------------------------------------
// Prep kernels
// ---------------------------------------------------------------------------

// Zero output (pad positions must be 0) and initial h/c state.
__global__ void k_init(float* __restrict__ out, float* __restrict__ h32,
                       float* __restrict__ c32, __bf16* __restrict__ hbf) {
  size_t i = (size_t)blockIdx.x * blockDim.x + threadIdx.x;
  const size_t NOUT = (size_t)T_STEPS * BATCH * HID;
  if (i < NOUT) out[i] = 0.0f;
  if (i < (size_t)2 * BATCH * HID) { h32[i] = 0.0f; c32[i] = 0.0f; }
  if (i < (size_t)4 * BATCH * HID) hbf[i] = (__bf16)0.0f;  // both parities, both dirs
}

// x (T,B,I) f32 -> bf16 copy (row-major, rows 16B-aligned).
__global__ void k_cvt_x(const float* __restrict__ x, __bf16* __restrict__ xb) {
  size_t i = (size_t)blockIdx.x * blockDim.x + threadIdx.x;
  if (i < (size_t)T_STEPS * BATCH * IN) xb[i] = (__bf16)x[i];
}

// Pack fused weights Wcat[dir][K=1024][N=2048] (K<512 -> w_ih[n][K], else
// w_hh[n][K-512]) into WMMA-B fragment order:
//   packedB[dir][kt][nt][lane][e], e=0..15, K = kt*32 + (lane>=16)*16 + e,
//   N = nt*16 + (lane&15).  32 bytes contiguous per lane per tile.
__global__ void k_pack_w(const float* __restrict__ wih_f, const float* __restrict__ whh_f,
                         const float* __restrict__ wih_b, const float* __restrict__ whh_b,
                         __bf16* __restrict__ packedB) {
  uint32_t tid = blockIdx.x * blockDim.x + threadIdx.x;
  if (tid >= 2u * KT * NT * 32 * 16) return;
  uint32_t e    = tid & 15u;
  uint32_t lane = (tid >> 4) & 31u;
  uint32_t nt   = (tid >> 9) & 127u;
  uint32_t kt   = (tid >> 16) & 31u;
  uint32_t d    = tid >> 21;
  uint32_t K = kt * 32u + ((lane >= 16u) ? 16u : 0u) + e;
  uint32_t n = nt * 16u + (lane & 15u);
  const float* wih = d ? wih_b : wih_f;
  const float* whh = d ? whh_b : whh_f;
  float v = (K < IN) ? wih[(size_t)n * IN + K] : whh[(size_t)n * HID + (K - IN)];
  packedB[tid] = (__bf16)v;
}

// bias[dir][4H] = b_ih + b_hh
__global__ void k_bias(const float* __restrict__ bi_f, const float* __restrict__ bh_f,
                       const float* __restrict__ bi_b, const float* __restrict__ bh_b,
                       float* __restrict__ bias) {
  int i = blockIdx.x * blockDim.x + threadIdx.x;
  if (i >= 2 * G4H) return;
  int d = i / G4H, n = i % G4H;
  bias[i] = d ? (bi_b[n] + bh_b[n]) : (bi_f[n] + bh_f[n]);
}

// ---------------------------------------------------------------------------
// One recurrence step, both directions.
// grid = 64 blocks: blockIdx.x = dir*32 + hb  (hb selects 16 hidden units)
// block = 64 threads = 2 waves: wave = batch M-tile (rows 0-15 / 16-31)
// Each wave accumulates the i/f/g/o 16x16 tiles for its hidden units fully
// in registers: 16 K-iterations x 4 WMMA per half, B staged in LDS by TDM.
// ---------------------------------------------------------------------------
__global__ __launch_bounds__(64)
void k_step(const __bf16* __restrict__ xb,       // (T,B,I) bf16
            const __bf16* __restrict__ packedB,  // [2][KT][NT][32][16] bf16
            const float*  __restrict__ bias,     // [2][4H]
            const int*    __restrict__ lengths,  // [B]
            const __bf16* __restrict__ h_in,     // parity p:   [2][B*H] bf16
            __bf16*       __restrict__ h_out,    // parity 1-p: [2][B*H] bf16
            float*        __restrict__ h32,      // [2][B*H] f32 master h
            float*        __restrict__ c32,      // [2][B*H] f32 cell state
            float*        __restrict__ out,      // (T,B,H) f32 (atomic add)
            int t) {
  // [half][kt][gate][512 bf16 fragment-packed tile] : 2 x 64 KB
  __shared__ __align__(16) __bf16 smemB[2][KTH * 4 * 512];

  const int lane = threadIdx.x & 31;
  const int mt   = threadIdx.x >> 5;      // batch M-tile (0,1)
  const int dir  = blockIdx.x >> 5;       // 0=fwd, 1=bwd
  const int hb   = blockIdx.x & 31;       // hidden-unit group (16 units)

  const __bf16* __restrict__ pB = packedB + (size_t)dir * ((size_t)KT * NT * 32 * 16);
  // Global start of this block's B tile chain: tile (kt=0, gate=0, hb).
  const __bf16* __restrict__ bstart = pB + (size_t)hb * 512;

#if HAVE_TDM
  if (threadIdx.x < 32) {  // wave 0 drives the DMA for the whole block
    tdm_load_B(bstart, (uint32_t)(uintptr_t)&smemB[0][0], KTH);
    tdm_load_B(bstart + (size_t)KTH * (NT * 512),
               (uint32_t)(uintptr_t)&smemB[1][0], KTH);
    __builtin_amdgcn_s_wait_tensorcnt(1);  // first half landed
  }
#else
  // Fallback: cooperative global->LDS copy of both halves (same layout).
  for (int i = threadIdx.x; i < 2 * KTH * 4 * 512 / 8; i += 64) {
    int chunk = i & 63;              // 16B chunk within a 1KB tile
    int tile  = i >> 6;              // (half*16 + kt)*4 + gate
    int g     = tile & 3;
    int kt    = tile >> 2;           // 0..31 across both halves
    const v8bf* src = (const v8bf*)(pB + ((size_t)kt * NT + g * 32 + hb) * 512) + chunk;
    ((v8bf*)&smemB[0][0])[i] = *src;
  }
#endif
  __syncthreads();

  // --- A-fragment addressing (contiguous K order, same as packed B):
  // lane L holds row M = L&15; elems 0..15 -> K = kb + hs + 0..15,
  // with hs = (L>=16)*16.
  const int bm = mt * 16 + (lane & 15);
  const int hs = (lane >= 16) ? 16 : 0;
  const int len_bm = lengths[bm];
  // Backward dir reads x[len-1-t] (clamped; frozen rows are masked out later).
  const int xt = dir ? (((len_bm - 1 - t) > 0) ? (len_bm - 1 - t) : 0) : t;

  const __bf16* __restrict__ xrow = xb + ((size_t)xt * BATCH + bm) * IN;
  const __bf16* __restrict__ hrow = h_in + (size_t)dir * (BATCH * HID) + (size_t)bm * HID;

  v8f acc0 = {};  // gate i
  v8f acc1 = {};  // gate f
  v8f acc2 = {};  // gate g
  v8f acc3 = {};  // gate o

#pragma unroll
  for (int half = 0; half < 2; ++half) {
    const __bf16* __restrict__ ab = half ? hrow : xrow;   // loop-invariant base
    const __bf16* __restrict__ sm = &smemB[half][0] + lane * 16;

#pragma unroll
    for (int kk = 0; kk < KTH; ++kk) {
      const __bf16* s = ab + kk * 32;
      v16bf a = cat16(*(const v8bf*)(s + hs), *(const v8bf*)(s + hs + 8));
      const __bf16* bt = sm + kk * (4 * 512);
      v16bf b0 = cat16(*(const v8bf*)(bt),         *(const v8bf*)(bt + 8));
      v16bf b1 = cat16(*(const v8bf*)(bt + 512),   *(const v8bf*)(bt + 512 + 8));
      v16bf b2 = cat16(*(const v8bf*)(bt + 1024),  *(const v8bf*)(bt + 1024 + 8));
      v16bf b3 = cat16(*(const v8bf*)(bt + 1536),  *(const v8bf*)(bt + 1536 + 8));
      acc0 = wmma_bf16(a, b0, acc0);
      acc1 = wmma_bf16(a, b1, acc1);
      acc2 = wmma_bf16(a, b2, acc2);
      acc3 = wmma_bf16(a, b3, acc3);
    }

#if HAVE_TDM
    if (half == 0) {
      if (threadIdx.x < 32) __builtin_amdgcn_s_wait_tensorcnt(0);  // 2nd half landed
      __syncthreads();
    }
#endif
  }

  // --- Element-wise LSTM cell, fully in registers.
  // C/D layout: VGPR r, lanes 0-15 -> M=r, lanes 16-31 -> M=8+r; N = lane&15.
  const int j = hb * 16 + (lane & 15);           // hidden unit
  const float bi = bias[dir * G4H + 0 * HID + j];
  const float bf = bias[dir * G4H + 1 * HID + j];
  const float bg = bias[dir * G4H + 2 * HID + j];
  const float bo = bias[dir * G4H + 3 * HID + j];
  const int rbase = mt * 16 + ((lane >= 16) ? 8 : 0);
  const size_t dbase = (size_t)dir * (BATCH * HID);

#pragma unroll
  for (int r = 0; r < 8; ++r) {
    const int b = rbase + r;
    const int len_b = lengths[b];
    const size_t sidx = dbase + (size_t)b * HID + j;
    if (t < len_b) {
      float iv = sigmoidf_(acc0[r] + bi);
      float fv = sigmoidf_(acc1[r] + bf);
      float gv = tanhf(acc2[r] + bg);
      float ov = sigmoidf_(acc3[r] + bo);
      float c_new = fv * c32[sidx] + iv * gv;
      float h_new = ov * tanhf(c_new);
      c32[sidx] = c_new;
      h32[sidx] = h_new;
      h_out[sidx] = (__bf16)h_new;
      const int to = dir ? (len_b - 1 - t) : t;
      atomicAdd(out + ((size_t)to * BATCH + b) * HID + j, h_new);
    } else {
      // Frozen past this sequence's length: carry h into the next parity.
      h_out[sidx] = h_in[sidx];
    }
  }
}

// h_n (2,B,H) and c_n (2,B,H) appended after out (T,B,H).
__global__ void k_final(const float* __restrict__ h32, const float* __restrict__ c32,
                        float* __restrict__ out) {
  int i = blockIdx.x * blockDim.x + threadIdx.x;
  const size_t NOUT = (size_t)T_STEPS * BATCH * HID;
  const int NS = 2 * BATCH * HID;
  if (i < NS) {
    out[NOUT + i] = h32[i];
    out[NOUT + NS + i] = c32[i];
  }
}

// ---------------------------------------------------------------------------
// Host launcher (graph-capture safe: only kernel launches on `stream`).
// ---------------------------------------------------------------------------
extern "C" void kernel_launch(void* const* d_in, const int* in_sizes, int n_in,
                              void* d_out, int out_size, void* d_ws, size_t ws_size,
                              hipStream_t stream) {
  const float* x     = (const float*)d_in[0];
  const int*   lens  = (const int*)  d_in[1];
  const float* wih_f = (const float*)d_in[2];
  const float* whh_f = (const float*)d_in[3];
  const float* bih_f = (const float*)d_in[4];
  const float* bhh_f = (const float*)d_in[5];
  const float* wih_b = (const float*)d_in[6];
  const float* whh_b = (const float*)d_in[7];
  const float* bih_b = (const float*)d_in[8];
  const float* bhh_b = (const float*)d_in[9];
  float* out = (float*)d_out;

  // Workspace carve-up (~48.2 MB total).
  char* ws = (char*)d_ws;
  size_t off = 0;
  __bf16* xb  = (__bf16*)(ws + off); off += (size_t)T_STEPS * BATCH * IN * 2;        // 39.3 MB
  __bf16* pB  = (__bf16*)(ws + off); off += (size_t)2 * KT * NT * 32 * 16 * 2;       //  8.4 MB
  float*  bia = (float*) (ws + off); off += (size_t)2 * G4H * 4;                     //  16 KB
  __bf16* hbf = (__bf16*)(ws + off); off += (size_t)2 * 2 * BATCH * HID * 2;         // 128 KB (parity x dir)
  float*  h32 = (float*) (ws + off); off += (size_t)2 * BATCH * HID * 4;             // 128 KB
  float*  c32 = (float*) (ws + off); off += (size_t)2 * BATCH * HID * 4;             // 128 KB
  (void)off; (void)ws_size; (void)in_sizes; (void)n_in; (void)out_size;

  const size_t NOUT = (size_t)T_STEPS * BATCH * HID;
  k_init<<<(unsigned)((NOUT + 255) / 256), 256, 0, stream>>>(out, h32, c32, hbf);
  k_cvt_x<<<(unsigned)(((size_t)T_STEPS * BATCH * IN + 255) / 256), 256, 0, stream>>>(x, xb);
  k_pack_w<<<(2u * KT * NT * 32 * 16 + 255) / 256, 256, 0, stream>>>(wih_f, whh_f, wih_b, whh_b, pB);
  k_bias<<<(2 * G4H + 255) / 256, 256, 0, stream>>>(bih_f, bhh_f, bih_b, bhh_b, bia);

  for (int t = 0; t < T_STEPS; ++t) {
    __bf16* hin  = hbf + (size_t)(t & 1)       * (2 * BATCH * HID);
    __bf16* hout = hbf + (size_t)((t + 1) & 1) * (2 * BATCH * HID);
    k_step<<<64, 64, 0, stream>>>(xb, pB, bia, lens, hin, hout, h32, c32, out, t);
  }

  k_final<<<(2 * BATCH * HID + 255) / 256, 256, 0, stream>>>(h32, c32, out);
}
